// KimiDeltaAttention_43963285242612
// MI455X (gfx1250) — matrix operand order repaired and checked by reading, not compiled, not run on hardware
//
#include <hip/hip_runtime.h>

// ---------------------------------------------------------------------------
// Types
// ---------------------------------------------------------------------------
typedef __bf16 bf16_t;
typedef __attribute__((ext_vector_type(16))) __bf16 v16bf;
typedef __attribute__((ext_vector_type(8)))  __bf16 v8bf;
typedef __attribute__((ext_vector_type(8)))  float  v8f;
typedef __attribute__((ext_vector_type(4)))  int    v4i;
typedef __attribute__((ext_vector_type(8)))  int    v8i;

#define BT   2048      // B*T
#define HIDW 1024
#define NH   8
#define HD   128

// ---------------------------------------------------------------------------
// fp32 -> bf16 conversion
// ---------------------------------------------------------------------------
__global__ void cvt_bf16(const float* __restrict__ in, bf16_t* __restrict__ out, int n) {
  int i = blockIdx.x * blockDim.x + threadIdx.x;
  if (i < n) out[i] = (bf16_t)in[i];
}

// ---------------------------------------------------------------------------
// WMMA bf16 GEMM:  C[M,N] (f32) = A[M,K] (bf16, row-major) x W[N,K]^T (bf16)
// Block tile 128x128, 8 waves, per-wave 64x32 = 8 WMMA tiles.
// Double-buffered LDS staging:
//   A tile (128x32) : per-lane GLOBAL_LOAD_ASYNC_TO_LDS_B128  (ASYNCcnt)
//   B tile (128x32) : one TENSOR_LOAD_TO_LDS descriptor, wave 0 (TENSORcnt)
// Steady-state loop is branch-free; final K-step is peeled for drain waits.
// ---------------------------------------------------------------------------
#define BM 128
#define BN 128
#define BK 32

// CDNA5 16-bit A-matrix 16x32 per-lane layout: lane m=L&15, kh=L>>4;
// halves hold K = [8kh, 8kh+8) and [16+8kh, 16+8kh+8).
__device__ __forceinline__ v16bf load_a_frag(const bf16_t* __restrict__ row, int kh) {
  const v8bf lo = *(const v8bf*)(row + 8 * kh);
  const v8bf hi = *(const v8bf*)(row + 16 + 8 * kh);
  v16bf r;
#pragma unroll
  for (int i = 0; i < 8; ++i) { r[i] = lo[i]; r[i + 8] = hi[i]; }
  return r;
}

// CDNA5 16-bit B-matrix 32x16 per-lane layout: lane n=L&15, kh=L>>4;
// halves hold K = [8kh, 8kh+8) and [16+8kh, 16+8kh+8).
__device__ __forceinline__ v16bf load_b_frag(const bf16_t* __restrict__ row, int kh) {
  const v8bf lo = *(const v8bf*)(row + 8 * kh);
  const v8bf hi = *(const v8bf*)(row + 16 + 8 * kh);
  v16bf r;
#pragma unroll
  for (int i = 0; i < 8; ++i) { r[i] = lo[i]; r[i + 8] = hi[i]; }
  return r;
}

// per-lane async copy of the A tile (128x32 bf16 = 512 x 16B, 2 chunks/thread)
__device__ __forceinline__ void stage_a_async(const bf16_t* __restrict__ A,
                                              bf16_t (*As)[BK],
                                              int bm, int K, int k0, int tid) {
#pragma unroll
  for (int i = 0; i < 2; ++i) {
    const int idx = tid + i * 256;
    const int r = idx >> 2, c = (idx & 3) * 8;
    const unsigned long long g =
        (unsigned long long)(size_t)&A[(size_t)(bm + r) * K + k0 + c];
    const unsigned lds = (unsigned)(size_t)&As[r][c];
    asm volatile("global_load_async_to_lds_b128 %0, %1, off"
                 :: "v"(lds), "v"(g) : "memory");
  }
}

// TDM descriptor (D# groups 0/1, 2-D tile) for the B tile: 128 rows x 32 cols
// of bf16, row stride K elements, contiguous rows in LDS.
__device__ __forceinline__ void stage_b_tdm(const bf16_t* __restrict__ W,
                                            unsigned ldsOff, int bn, int K, int k0) {
  const unsigned long long ga =
      (unsigned long long)(size_t)(W + (size_t)bn * K + k0);
  v4i g0; v8i g1;
  g0[0] = 1;                                          // count=1, user mode
  g0[1] = (int)ldsOff;                                // lds_addr
  g0[2] = (int)(ga & 0xffffffffu);                    // global_addr[31:0]
  g0[3] = (int)((ga >> 32) & 0x1ffffff) | (2 << 30);  // addr[56:32] | type=2
  g1[0] = (1 << 16);          // workgroup_mask=0 | data_size=1 (2 bytes)
  g1[1] = (BK << 16);         // atomic_barrier=0 | tensor_dim0[15:0]=32
  g1[2] = (BN << 16);         // tensor_dim0 hi=0 | tensor_dim1[15:0]=128
  g1[3] = (BK << 16);         // tensor_dim1 hi=0 | tile_dim0=32
  g1[4] = BN;                 // tile_dim1=128 | tile_dim2=0 (2-D)
  g1[5] = K;                  // tensor_dim0_stride[31:0]
  g1[6] = 0;                  // stride0 hi | stride1 lo (unused, 2-D)
  g1[7] = 0;
  asm volatile("tensor_load_to_lds %0, %1" :: "s"(g0), "s"(g1) : "memory");
}

// one K-step of WMMA math on the staged buffers (8 x v_wmma per wave)
__device__ __forceinline__ void wmma_step(const bf16_t (*As)[BK], const bf16_t (*Bs)[BK],
                                          v8f (&acc)[4][2], int wm, int wn,
                                          int mrow, int kh) {
  v16bf afr[4], bfr[2];
#pragma unroll
  for (int mt = 0; mt < 4; ++mt)
    afr[mt] = load_a_frag(&As[wm + mt * 16 + mrow][0], kh);
#pragma unroll
  for (int nt = 0; nt < 2; ++nt)
    bfr[nt] = load_b_frag(&Bs[wn + nt * 16 + mrow][0], kh);
#pragma unroll
  for (int mt = 0; mt < 4; ++mt)
#pragma unroll
    for (int nt = 0; nt < 2; ++nt)
      acc[mt][nt] = __builtin_amdgcn_wmma_f32_16x16x32_bf16(
          false, afr[mt], false, bfr[nt], (short)0, acc[mt][nt], false, false);
}

__global__ __launch_bounds__(256)
void gemm_bf16_nt(const bf16_t* __restrict__ A, const bf16_t* __restrict__ W,
                  float* __restrict__ C, int M, int N, int K) {
  __shared__ __align__(16) bf16_t As[2][BM][BK];   // 2 x 8 KB
  __shared__ __align__(16) bf16_t Bs[2][BN][BK];   // 2 x 8 KB

  const int tid  = threadIdx.x;
  const int lane = tid & 31;
  const int wave = tid >> 5;
  const int bm = blockIdx.y * BM;
  const int bn = blockIdx.x * BN;
  const int wm = (wave & 1) * 64;       // wave grid 2 (M) x 4 (N), 64x32/wave
  const int wn = (wave >> 1) * 32;
  const int mrow = lane & 15;
  const int kh   = lane >> 4;

  v8f acc[4][2] = {};

  // prologue: stage buffer 0
  stage_a_async(A, As[0], bm, K, 0, tid);
  if (wave == 0) stage_b_tdm(W, (unsigned)(size_t)&Bs[0][0][0], bn, K, 0);

  const int nsteps = K / BK;
  int cur = 0;
  // steady state: branch-free issue-next / wait / compute
  for (int s = 0; s < nsteps - 1; ++s) {
    const int k0 = (s + 1) * BK;
    stage_a_async(A, As[cur ^ 1], bm, K, k0, tid);
    if (wave == 0)
      stage_b_tdm(W, (unsigned)(size_t)&Bs[cur ^ 1][0][0], bn, K, k0);
    asm volatile("s_wait_asynccnt 0x2" ::: "memory");  // leave next-buf inflight
    __builtin_amdgcn_s_wait_tensorcnt((short)1);       // no-op on waves 1..7
    __syncthreads();        // staged tile visible to all waves
    wmma_step(As[cur], Bs[cur], acc, wm, wn, mrow, kh);
    __syncthreads();        // all reads of `cur` done before it is restaged
    cur ^= 1;
  }
  // peeled final K-step: drain all outstanding copies
  asm volatile("s_wait_asynccnt 0x0" ::: "memory");
  __builtin_amdgcn_s_wait_tensorcnt((short)0);
  __syncthreads();
  wmma_step(As[cur], Bs[cur], acc, wm, wn, mrow, kh);

  // C/D f32 16x16 layout: VGPR r -> M = r + 8*kh, N = lane&15
#pragma unroll
  for (int mt = 0; mt < 4; ++mt)
#pragma unroll
    for (int nt = 0; nt < 2; ++nt) {
      const int col = bn + wn + nt * 16 + mrow;
#pragma unroll
      for (int r = 0; r < 8; ++r) {
        const int row = bm + wm + mt * 16 + 8 * kh + r;
        C[(size_t)row * N + col] = acc[mt][nt][r];
      }
    }
}

// ---------------------------------------------------------------------------
// beta = sigmoid(x @ Wb^T)   (N=8 -> plain f32 dot kernel, trivial cost)
// ---------------------------------------------------------------------------
__global__ void beta_kernel(const float* __restrict__ x, const float* __restrict__ Wb,
                            float* __restrict__ beta, int rows, int K) {
  const int i = blockIdx.x * blockDim.x + threadIdx.x;
  if (i >= rows * NH) return;
  const int row = i >> 3, h = i & 7;
  const float* xr = x + (size_t)row * K;
  const float* wr = Wb + (size_t)h * K;
  float s = 0.f;
  for (int k = 0; k < K; ++k) s = fmaf(xr[k], wr[k], s);
  beta[i] = 1.f / (1.f + __expf(-s));
}

// ---------------------------------------------------------------------------
// depthwise causal conv (K=4) + SiLU over (B,T,C) with C=1024, T=1024
// ---------------------------------------------------------------------------
__global__ void conv_silu(const float* __restrict__ in, const float* __restrict__ w,
                          float* __restrict__ out, int n) {
  const int idx = blockIdx.x * blockDim.x + threadIdx.x;
  if (idx >= n) return;
  const int c = idx & 1023;
  const int t = (idx >> 10) & 1023;
  float acc = 0.f;
#pragma unroll
  for (int j = 0; j < 4; ++j) {
    const int tt = t - 3 + j;
    if (tt >= 0) acc = fmaf(w[c * 4 + j], in[idx + (j - 3) * 1024], acc);
  }
  out[idx] = acc / (1.f + __expf(-acc));   // SiLU
}

// ---------------------------------------------------------------------------
// per-(b,t,h) RMS norm over D=128 with scalar multiplier (wave per row)
// ---------------------------------------------------------------------------
__global__ __launch_bounds__(256)
void rmsnorm_rows(float* __restrict__ p, float mul, float eps, int rows) {
  const int wid  = blockIdx.x * (blockDim.x >> 5) + (threadIdx.x >> 5);
  const int lane = threadIdx.x & 31;
  if (wid >= rows) return;
  float* r = p + (size_t)wid * HD;
  float v[4]; float ss = 0.f;
#pragma unroll
  for (int i = 0; i < 4; ++i) { v[i] = r[lane + 32 * i]; ss = fmaf(v[i], v[i], ss); }
#pragma unroll
  for (int off = 16; off; off >>= 1) ss += __shfl_xor(ss, off, 32);
  const float sc = rsqrtf(ss * (1.f / 128.f) + eps) * mul;
#pragma unroll
  for (int i = 0; i < 4; ++i) r[lane + 32 * i] = v[i] * sc;
}

// ---------------------------------------------------------------------------
// g = -exp(A_log[h]) * softplus(a_logits + dt_bias[h,d])   (in-place)
// ---------------------------------------------------------------------------
__global__ void g_kernel(float* __restrict__ a, const float* __restrict__ dt_bias,
                         const float* __restrict__ A_log, int n) {
  const int idx = blockIdx.x * blockDim.x + threadIdx.x;
  if (idx >= n) return;
  const int d = idx & 127;
  const int h = (idx >> 7) & 7;
  const float xv = a[idx] + dt_bias[h * HD + d];
  const float sp = (xv > 20.f) ? xv : log1pf(__expf(xv));
  a[idx] = -__expf(A_log[h]) * sp;
}

// ---------------------------------------------------------------------------
// gated delta-rule recurrence.  One block per (b,h); 256 threads; thread
// (half,dv) owns S[dk0..dk0+63][dv] in 64 VGPRs. Sequential over T.
// ---------------------------------------------------------------------------
__global__ __launch_bounds__(256)
void delta_recurrence(const float* __restrict__ q, const float* __restrict__ k,
                      const float* __restrict__ v, const float* __restrict__ g,
                      const float* __restrict__ beta, float* __restrict__ o, int T) {
  const int b = blockIdx.x >> 3, h = blockIdx.x & 7;
  const int tid  = threadIdx.x;
  const int dvi  = tid & 127;
  const int half = tid >> 7;          // 0: dk in [0,64), 1: dk in [64,128)
  const int dk0  = half << 6;

  __shared__ float qs[HD], ks[HD], eg[HD], vs[HD];
  __shared__ float kvp[256], opart[256];

  float S[64];
#pragma unroll
  for (int i = 0; i < 64; ++i) S[i] = 0.f;

  for (int t = 0; t < T; ++t) {
    const size_t base = ((size_t)((b * T + t) * NH + h)) << 7;
    if (half == 0) { qs[dvi] = q[base + dvi]; ks[dvi] = k[base + dvi]; }
    else           { eg[dvi] = __expf(g[base + dvi]); vs[dvi] = v[base + dvi]; }
    const float bt = beta[(size_t)(b * T + t) * NH + h];
    __syncthreads();

    float kv = 0.f;
#pragma unroll
    for (int i = 0; i < 64; ++i) {        // decay + read
      const float s = S[i] * eg[dk0 + i];
      S[i] = s;
      kv = fmaf(ks[dk0 + i], s, kv);
    }
    kvp[tid] = kv;
    __syncthreads();

    const float delta = bt * (vs[dvi] - (kvp[dvi] + kvp[dvi + 128]));
    float op = 0.f;
#pragma unroll
    for (int i = 0; i < 64; ++i) {        // rank-1 write + output read
      const float s = fmaf(ks[dk0 + i], delta, S[i]);
      S[i] = s;
      op = fmaf(qs[dk0 + i], s, op);
    }
    opart[tid] = op;
    __syncthreads();

    if (half == 0) o[base + dvi] = opart[dvi] + opart[dvi + 128];
    __syncthreads();
  }
}

// ---------------------------------------------------------------------------
// out = rmsnorm(o) * o_norm_w * sigmoid(gate)   (in-place, wave per row)
// ---------------------------------------------------------------------------
__global__ __launch_bounds__(256)
void out_norm_gate(float* __restrict__ o, const float* __restrict__ gate,
                   const float* __restrict__ w, int rows) {
  const int wid  = blockIdx.x * (blockDim.x >> 5) + (threadIdx.x >> 5);
  const int lane = threadIdx.x & 31;
  if (wid >= rows) return;
  float* p = o + (size_t)wid * HD;
  const float* gp = gate + (size_t)wid * HD;
  float v[4]; float ss = 0.f;
#pragma unroll
  for (int i = 0; i < 4; ++i) { v[i] = p[lane + 32 * i]; ss = fmaf(v[i], v[i], ss); }
#pragma unroll
  for (int off = 16; off; off >>= 1) ss += __shfl_xor(ss, off, 32);
  const float r = rsqrtf(ss * (1.f / 128.f) + 1e-5f);
#pragma unroll
  for (int i = 0; i < 4; ++i) {
    const int d = lane + 32 * i;
    const float gv = gp[d];
    p[d] = v[i] * r * w[d] * (1.f / (1.f + __expf(-gv)));
  }
}

// ---------------------------------------------------------------------------
// launcher
// ---------------------------------------------------------------------------
extern "C" void kernel_launch(void* const* d_in, const int* in_sizes, int n_in,
                              void* d_out, int out_size, void* d_ws, size_t ws_size,
                              hipStream_t stream) {
  (void)in_sizes; (void)n_in; (void)out_size; (void)ws_size;

  const float* x       = (const float*)d_in[0];
  const float* Wq      = (const float*)d_in[1];
  const float* Wk      = (const float*)d_in[2];
  const float* Wv      = (const float*)d_in[3];
  const float* wq_conv = (const float*)d_in[4];
  const float* wk_conv = (const float*)d_in[5];
  const float* wv_conv = (const float*)d_in[6];
  const float* Wfa     = (const float*)d_in[7];
  const float* Wfb     = (const float*)d_in[8];
  const float* Wb      = (const float*)d_in[9];
  const float* Wga     = (const float*)d_in[10];
  const float* Wgb     = (const float*)d_in[11];
  const float* A_log   = (const float*)d_in[12];
  const float* dt_bias = (const float*)d_in[13];
  const float* o_nw    = (const float*)d_in[14];
  const float* Wo      = (const float*)d_in[15];

  // workspace carve-up (256B aligned)
  char* base = (char*)d_ws;
  size_t off = 0;
  auto take = [&](size_t bytes) -> void* {
    void* p = base + off;
    off = (off + bytes + 255) & ~(size_t)255;
    return p;
  };
  bf16_t* xb   = (bf16_t*)take((size_t)BT * HIDW * 2);
  bf16_t* wqb  = (bf16_t*)take((size_t)HIDW * HIDW * 2);
  bf16_t* wkb  = (bf16_t*)take((size_t)HIDW * HIDW * 2);
  bf16_t* wvb  = (bf16_t*)take((size_t)HIDW * HIDW * 2);
  bf16_t* wob  = (bf16_t*)take((size_t)HIDW * HIDW * 2);
  bf16_t* wfab = (bf16_t*)take((size_t)HD * HIDW * 2);
  bf16_t* wfbb = (bf16_t*)take((size_t)HIDW * HD * 2);
  bf16_t* wgab = (bf16_t*)take((size_t)HD * HIDW * 2);
  bf16_t* wgbb = (bf16_t*)take((size_t)HIDW * HD * 2);
  float*  P0   = (float*)take((size_t)BT * HIDW * 4);   // q_pre -> a_logits/g
  float*  P1   = (float*)take((size_t)BT * HIDW * 4);   // k_pre -> gate
  float*  P2   = (float*)take((size_t)BT * HIDW * 4);   // v_pre -> o
  float*  C0   = (float*)take((size_t)BT * HIDW * 4);   // q (conv+norm)
  float*  C1   = (float*)take((size_t)BT * HIDW * 4);   // k
  float*  C2   = (float*)take((size_t)BT * HIDW * 4);   // v
  float*  fa   = (float*)take((size_t)BT * HD * 4);
  float*  ga   = (float*)take((size_t)BT * HD * 4);
  bf16_t* fab  = (bf16_t*)take((size_t)BT * HD * 2);
  bf16_t* gab  = (bf16_t*)take((size_t)BT * HD * 2);
  float*  betb = (float*)take((size_t)BT * NH * 4);
  bf16_t* ob   = (bf16_t*)take((size_t)BT * HIDW * 2);

  auto g1 = [](int n) { return (n + 255) / 256; };
  const int nBig = BT * HIDW;        // 2M
  const int nW   = HIDW * HIDW;      // 1M
  const int nLR  = HD * HIDW;        // 128K

  // 1) precision staging
  cvt_bf16<<<g1(nBig), 256, 0, stream>>>(x,   xb,   nBig);
  cvt_bf16<<<g1(nW),   256, 0, stream>>>(Wq,  wqb,  nW);
  cvt_bf16<<<g1(nW),   256, 0, stream>>>(Wk,  wkb,  nW);
  cvt_bf16<<<g1(nW),   256, 0, stream>>>(Wv,  wvb,  nW);
  cvt_bf16<<<g1(nW),   256, 0, stream>>>(Wo,  wob,  nW);
  cvt_bf16<<<g1(nLR),  256, 0, stream>>>(Wfa, wfab, nLR);
  cvt_bf16<<<g1(nLR),  256, 0, stream>>>(Wfb, wfbb, nLR);
  cvt_bf16<<<g1(nLR),  256, 0, stream>>>(Wga, wgab, nLR);
  cvt_bf16<<<g1(nLR),  256, 0, stream>>>(Wgb, wgbb, nLR);

  // 2) projections (WMMA, async+TDM staged)
  dim3 gq(HIDW / BN, BT / BM);       // (8, 16)
  dim3 gf(1, BT / BM);               // (1, 16)
  gemm_bf16_nt<<<gq, 256, 0, stream>>>(xb, wqb,  P0, BT, HIDW, HIDW);
  gemm_bf16_nt<<<gq, 256, 0, stream>>>(xb, wkb,  P1, BT, HIDW, HIDW);
  gemm_bf16_nt<<<gq, 256, 0, stream>>>(xb, wvb,  P2, BT, HIDW, HIDW);
  gemm_bf16_nt<<<gf, 256, 0, stream>>>(xb, wfab, fa, BT, HD,   HIDW);
  gemm_bf16_nt<<<gf, 256, 0, stream>>>(xb, wgab, ga, BT, HD,   HIDW);
  beta_kernel<<<g1(BT * NH), 256, 0, stream>>>(x, Wb, betb, BT, HIDW);

  // 3) short conv + SiLU
  conv_silu<<<g1(nBig), 256, 0, stream>>>(P0, wq_conv, C0, nBig);
  conv_silu<<<g1(nBig), 256, 0, stream>>>(P1, wk_conv, C1, nBig);
  conv_silu<<<g1(nBig), 256, 0, stream>>>(P2, wv_conv, C2, nBig);

  // 4) q/k RMS norms: q *= scale^2 = 1/D, k *= scale = D^-0.5
  rmsnorm_rows<<<(BT * NH) / 8, 256, 0, stream>>>(C0, 1.f / 128.f, 1e-6f, BT * NH);
  rmsnorm_rows<<<(BT * NH) / 8, 256, 0, stream>>>(C1, 0.0883883476483184f, 1e-6f, BT * NH);

  // 5) low-rank a_logits / gate (WMMA, K=128)
  cvt_bf16<<<g1(BT * HD), 256, 0, stream>>>(fa, fab, BT * HD);
  cvt_bf16<<<g1(BT * HD), 256, 0, stream>>>(ga, gab, BT * HD);
  gemm_bf16_nt<<<gq, 256, 0, stream>>>(fab, wfbb, P0, BT, HIDW, HD);   // a_logits
  gemm_bf16_nt<<<gq, 256, 0, stream>>>(gab, wgbb, P1, BT, HIDW, HD);   // gate

  // 6) g = -exp(A_log)*softplus(a_logits + dt_bias)  (in-place in P0)
  g_kernel<<<g1(nBig), 256, 0, stream>>>(P0, dt_bias, A_log, nBig);

  // 7) sequential gated delta-rule scan; o -> P2
  delta_recurrence<<<2 * NH, 256, 0, stream>>>(C0, C1, C2, P0, betb, P2, 1024);

  // 8) output norm * o_norm_w * sigmoid(gate)  (in-place in P2)
  out_norm_gate<<<(BT * NH) / 8, 256, 0, stream>>>(P2, P1, o_nw, BT * NH);

  // 9) final projection (WMMA) into d_out
  cvt_bf16<<<g1(nBig), 256, 0, stream>>>(P2, ob, nBig);
  gemm_bf16_nt<<<gq, 256, 0, stream>>>(ob, wob, (float*)d_out, BT, HIDW, HIDW);
}
